// RibonucleicAcidSE3Transformer_35527969472691
// MI455X (gfx1250) — matrix operand order repaired and checked by reading, not compiled, not run on hardware
//
#include <hip/hip_runtime.h>
#include <math.h>

#define B_    2
#define N_    2048
#define K_    32
#define H_    64
#define CV_   16
#define L_    4
#define DIN_  176
#define DINP_ 192
#define NODES_ (B_*N_)
#define E_     (B_*N_*K_)

typedef __attribute__((ext_vector_type(16))) _Float16 v16h;
typedef __attribute__((ext_vector_type(8)))  _Float16 v8h;
typedef __attribute__((ext_vector_type(2)))  _Float16 v2h;
typedef __attribute__((ext_vector_type(8)))  float    v8f;

union F16Frag { v16h v; v8h h[2]; };

#define WMMA_F16(A,Bf,C) __builtin_amdgcn_wmma_f32_16x16x32_f16(false,(A),false,(Bf),(short)0,(C),false,false)

// ---------------- coordinate centering ----------------
__global__ void k_center(const float* __restrict__ coord, float* __restrict__ coordC) {
  __shared__ float sx[256], sy[256], sz[256];
  int b = blockIdx.x, t = threadIdx.x;
  float ax = 0.f, ay = 0.f, az = 0.f;
  for (int i = t; i < N_; i += 256) {
    int g = (b*N_ + i)*3;
    ax += coord[g]; ay += coord[g+1]; az += coord[g+2];
  }
  sx[t] = ax; sy[t] = ay; sz[t] = az;
  __syncthreads();
  for (int s = 128; s > 0; s >>= 1) {
    if (t < s) { sx[t] += sx[t+s]; sy[t] += sy[t+s]; sz[t] += sz[t+s]; }
    __syncthreads();
  }
  float mx = sx[0]*(1.f/N_), my = sy[0]*(1.f/N_), mz = sz[0]*(1.f/N_);
  for (int i = t; i < N_; i += 256) {
    int g = (b*N_ + i)*3;
    coordC[g]   = coord[g]   - mx;
    coordC[g+1] = coord[g+1] - my;
    coordC[g+2] = coord[g+2] - mz;
  }
}

// ---------------- KNN: wave per destination row ----------------
__global__ __launch_bounds__(128) void k_knn(const float* __restrict__ coordC,
                                             int* __restrict__ idxOut) {
  __shared__ float sD[4][N_];
  int wv = threadIdx.x >> 5, lane = threadIdx.x & 31;
  int row = blockIdx.x*4 + wv;          // flattened b*N_+i
  int b = row / N_, i = row - b*N_;
  float cx = coordC[row*3], cy = coordC[row*3+1], cz = coordC[row*3+2];
  for (int j = lane; j < N_; j += 32) {
    int g = (b*N_ + j)*3;
    float dx = coordC[g] - cx, dy = coordC[g+1] - cy, dz = coordC[g+2] - cz;
    float d2 = dx*dx + dy*dy + dz*dz;
    if (j == i) d2 = 1e9f;
    sD[wv][j] = d2;
  }
  __syncthreads();
  for (int r = 0; r < K_; ++r) {
    float mval = 1e30f; int mj = N_;
    for (int j = lane; j < N_; j += 32) {
      float v = sD[wv][j];
      if (v < mval || (v == mval && j < mj)) { mval = v; mj = j; }
    }
    for (int off = 16; off > 0; off >>= 1) {
      float ov = __shfl_xor(mval, off);
      int   oj = __shfl_xor(mj, off);
      if (ov < mval || (ov == mval && oj < mj)) { mval = ov; mj = oj; }
    }
    if ((mj & 31) == lane) sD[wv][mj] = 1e30f;
    if (lane == 0) idxOut[row*K_ + r] = mj;
    __syncthreads();
  }
}

// ---------------- per-edge geometry: dirn + RBF ----------------
__global__ void k_edges(const float* __restrict__ coordC, const int* __restrict__ idxA,
                        float* __restrict__ dirn, float* __restrict__ rbf) {
  int e = blockIdx.x*blockDim.x + threadIdx.x;
  if (e >= E_) return;
  int node = e >> 5;            // /K_
  int b = node / N_;
  int src = idxA[e];
  const float* cd = coordC + node*3;
  const float* cs = coordC + (b*N_ + src)*3;
  float rx = cs[0]-cd[0], ry = cs[1]-cd[1], rz = cs[2]-cd[2];
  float d = sqrtf(rx*rx + ry*ry + rz*rz + 1e-8f);
  float inv = 1.f/d;
  dirn[e*3]   = rx*inv;
  dirn[e*3+1] = ry*inv;
  dirn[e*3+2] = rz*inv;
  const float twos2 = 2.f*0.625f*0.625f;   // 2*sigma^2, sigma = 10/16
  #pragma unroll
  for (int j = 0; j < 16; ++j) {
    float mu = 10.f*(float)j/15.f;
    float df = d - mu;
    rbf[e*16 + j] = expf(-df*df/twos2);
  }
}

// ---------------- f0 init: concat(atom,time) @ Wp0 + bp0 ----------------
__global__ void k_f0init(const int* __restrict__ seq, const int* __restrict__ tptr,
                         const float* __restrict__ atomT, const float* __restrict__ timeT,
                         const float* __restrict__ Wp0, const float* __restrict__ bp0,
                         float* __restrict__ f0) {
  int u = blockIdx.x*blockDim.x + threadIdx.x;   // < NODES_*H_
  int node = u >> 6, c = u & 63;
  int sv = seq[node];
  int t = *tptr;
  float acc = bp0[c];
  #pragma unroll 8
  for (int k = 0; k < 64; ++k) {
    float in = (k < 32) ? atomT[sv*32 + k] : timeT[t*32 + (k-32)];
    acc += in * Wp0[k*64 + c];
  }
  f0[u] = acc;
}

// ---------------- f1 init ----------------
__global__ void k_f1init(const float* __restrict__ coordC, const float* __restrict__ Wp1,
                         float* __restrict__ f1) {
  int u = blockIdx.x*blockDim.x + threadIdx.x;   // < NODES_*CV_*3
  int node = u / 48;
  int rem = u - node*48;
  int c = rem / 3, d = rem - c*3;
  f1[u] = coordC[node*3 + d] * Wp1[c];
}

// ---------------- per-layer: invariant dots <f1, dir> ----------------
__global__ void k_dots(const float* __restrict__ f1, const int* __restrict__ idxA,
                       const float* __restrict__ dirn, float* __restrict__ dots) {
  int e = blockIdx.x*blockDim.x + threadIdx.x;
  if (e >= E_) return;
  int node = e >> 5;
  int b = node / N_;
  int src = idxA[e];
  const float* fs = f1 + (b*N_ + src)*48;
  const float* fd = f1 + node*48;
  float dx = dirn[e*3], dy = dirn[e*3+1], dz = dirn[e*3+2];
  #pragma unroll
  for (int c = 0; c < 16; ++c) {
    dots[e*32 + c]      = fs[c*3]*dx + fs[c*3+1]*dy + fs[c*3+2]*dz;
    dots[e*32 + 16 + c] = fd[c*3]*dx + fd[c*3+1]*dy + fd[c*3+2]*dz;
  }
}

// ---------------- per-layer: edge MLP via WMMA (wave = 16-edge tile) ----------------
// LDS layouts chosen so every WMMA fragment is exactly two ds_load_b128:
//   A tile rows:  [16][192] f16, 16B aligned -> per lane two contiguous 16B runs
//   B weights:    stored TRANSPOSED [col][k] -> 16 contiguous K-halves per lane
__global__ __launch_bounds__(128) void k_edge_mlp(
    const float* __restrict__ f0, const int* __restrict__ idxA,
    const float* __restrict__ rbf, const float* __restrict__ dots,
    const float* __restrict__ We1, const float* __restrict__ be1,
    const float* __restrict__ We2, const float* __restrict__ be2,
    float* __restrict__ mOut, int l) {
  __shared__ __align__(16) _Float16 sWe1T[H_*DINP_];   // [col][k] 64x192 = 24 KB
  __shared__ __align__(16) _Float16 sWe2T[H_*H_];      // [col][k] 64x64  =  8 KB
  __shared__ __align__(16) _Float16 sE[4][16*DINP_];   // 4 tiles x 6 KB  = 24 KB
  const float* We1l = We1 + l*DIN_*H_;
  const float* We2l = We2 + l*H_*H_;
  const float* be1l = be1 + l*H_;
  const float* be2l = be2 + l*H_;

  // stage transposed weights (coalesced global reads, scattered u16 LDS writes)
  for (int u = threadIdx.x; u < DIN_*H_; u += 128) {
    int k = u >> 6, c = u & 63;
    sWe1T[c*DINP_ + k] = (_Float16)We1l[u];
  }
  for (int u = threadIdx.x; u < H_*(DINP_-DIN_); u += 128) {   // zero K-padding
    int c = u >> 4, k = DIN_ + (u & 15);
    sWe1T[c*DINP_ + k] = (_Float16)0.f;
  }
  for (int u = threadIdx.x; u < H_*H_; u += 128) {
    int k = u >> 6, c = u & 63;
    sWe2T[c*H_ + k] = (_Float16)We2l[u];
  }
  __syncthreads();

  int wv = threadIdx.x >> 5, lane = threadIdx.x & 31;
  int lh = lane >> 4, l15 = lane & 15;
  int tile = blockIdx.x*4 + wv;
  int eBase = tile*16;
  int node = eBase >> 5;        // all 16 edges share one dst node
  int bat = node / N_;
  _Float16* Et = &sE[wv][0];

  // stage e[16 x 192] as packed f16 pairs: [f0_dst | f0_src | rbf | dot_s | dot_d | pad]
  for (int u = lane; u < 16*(DINP_/2); u += 32) {
    int r = u / (DINP_/2);
    int c = (u - r*(DINP_/2))*2;
    int e = eBase + r;
    float x0, x1;
    if (c < 64)        { const float* p = f0 + node*64 + c;                    x0 = p[0]; x1 = p[1]; }
    else if (c < 128)  { int src = idxA[e];
                         const float* p = f0 + (bat*N_ + src)*64 + (c-64);     x0 = p[0]; x1 = p[1]; }
    else if (c < 144)  { const float* p = rbf + e*16 + (c-128);                x0 = p[0]; x1 = p[1]; }
    else if (c < DIN_) { const float* p = dots + e*32 + (c-144);               x0 = p[0]; x1 = p[1]; }
    else               { x0 = 0.f; x1 = 0.f; }
    v2h pk; pk[0] = (_Float16)x0; pk[1] = (_Float16)x1;
    *(v2h*)&Et[r*DINP_ + c] = pk;
  }
  __syncthreads();

  // GEMM1: h[16,64] = relu(e @ We1 + be1)
  v8f acc0, acc1, acc2, acc3;
  {
    float b0 = be1l[l15], b1 = be1l[16+l15], b2 = be1l[32+l15], b3 = be1l[48+l15];
    #pragma unroll
    for (int r = 0; r < 8; ++r) { acc0[r]=b0; acc1[r]=b1; acc2[r]=b2; acc3[r]=b3; }
  }
  #pragma unroll
  for (int kc = 0; kc < 6; ++kc) {
    F16Frag a;
    const _Float16* ar = &Et[l15*DINP_ + kc*32 + 16*lh];
    a.h[0] = *(const v8h*)ar;
    a.h[1] = *(const v8h*)(ar + 8);
    F16Frag bf;
    const _Float16* b0p = &sWe1T[l15*DINP_ + kc*32 + 16*lh];
    bf.h[0] = *(const v8h*)b0p;  bf.h[1] = *(const v8h*)(b0p + 8);
    acc0 = WMMA_F16(a.v, bf.v, acc0);
    const _Float16* b1p = &sWe1T[(16+l15)*DINP_ + kc*32 + 16*lh];
    bf.h[0] = *(const v8h*)b1p;  bf.h[1] = *(const v8h*)(b1p + 8);
    acc1 = WMMA_F16(a.v, bf.v, acc1);
    const _Float16* b2p = &sWe1T[(32+l15)*DINP_ + kc*32 + 16*lh];
    bf.h[0] = *(const v8h*)b2p;  bf.h[1] = *(const v8h*)(b2p + 8);
    acc2 = WMMA_F16(a.v, bf.v, acc2);
    const _Float16* b3p = &sWe1T[(48+l15)*DINP_ + kc*32 + 16*lh];
    bf.h[0] = *(const v8h*)b3p;  bf.h[1] = *(const v8h*)(b3p + 8);
    acc3 = WMMA_F16(a.v, bf.v, acc3);
  }
  __syncthreads();
  // relu; write h back into first 64 cols of the tile (D-layout scatter)
  #pragma unroll
  for (int r = 0; r < 8; ++r) {
    int mr = 8*lh + r;
    Et[mr*DINP_ + l15]      = (_Float16)fmaxf(acc0[r], 0.f);
    Et[mr*DINP_ + 16 + l15] = (_Float16)fmaxf(acc1[r], 0.f);
    Et[mr*DINP_ + 32 + l15] = (_Float16)fmaxf(acc2[r], 0.f);
    Et[mr*DINP_ + 48 + l15] = (_Float16)fmaxf(acc3[r], 0.f);
  }
  __syncthreads();

  // GEMM2: m[16,64] = h @ We2 + be2
  v8f c0, c1, c2, c3;
  {
    float b0 = be2l[l15], b1 = be2l[16+l15], b2 = be2l[32+l15], b3 = be2l[48+l15];
    #pragma unroll
    for (int r = 0; r < 8; ++r) { c0[r]=b0; c1[r]=b1; c2[r]=b2; c3[r]=b3; }
  }
  #pragma unroll
  for (int kc = 0; kc < 2; ++kc) {
    F16Frag a;
    const _Float16* ar = &Et[l15*DINP_ + kc*32 + 16*lh];
    a.h[0] = *(const v8h*)ar;
    a.h[1] = *(const v8h*)(ar + 8);
    F16Frag bf;
    const _Float16* b0p = &sWe2T[l15*H_ + kc*32 + 16*lh];
    bf.h[0] = *(const v8h*)b0p;  bf.h[1] = *(const v8h*)(b0p + 8);
    c0 = WMMA_F16(a.v, bf.v, c0);
    const _Float16* b1p = &sWe2T[(16+l15)*H_ + kc*32 + 16*lh];
    bf.h[0] = *(const v8h*)b1p;  bf.h[1] = *(const v8h*)(b1p + 8);
    c1 = WMMA_F16(a.v, bf.v, c1);
    const _Float16* b2p = &sWe2T[(32+l15)*H_ + kc*32 + 16*lh];
    bf.h[0] = *(const v8h*)b2p;  bf.h[1] = *(const v8h*)(b2p + 8);
    c2 = WMMA_F16(a.v, bf.v, c2);
    const _Float16* b3p = &sWe2T[(48+l15)*H_ + kc*32 + 16*lh];
    bf.h[0] = *(const v8h*)b3p;  bf.h[1] = *(const v8h*)(b3p + 8);
    c3 = WMMA_F16(a.v, bf.v, c3);
  }
  #pragma unroll
  for (int r = 0; r < 8; ++r) {
    int er = eBase + 8*lh + r;
    mOut[er*64 + l15]      = c0[r];
    mOut[er*64 + 16 + l15] = c1[r];
    mOut[er*64 + 32 + l15] = c2[r];
    mOut[er*64 + 48 + l15] = c3[r];
  }
}

// ---------------- per-layer: attention + f0/f1 update (wave = node) ----------------
__global__ __launch_bounds__(128) void k_node_update(
    const float* __restrict__ f0in, const float* __restrict__ f1in,
    const float* __restrict__ mArr, const int* __restrict__ idxA,
    const float* __restrict__ dirn,
    const float* __restrict__ Wq, const float* __restrict__ Wk,
    const float* __restrict__ Wv0, const float* __restrict__ Wv1,
    const float* __restrict__ Wg1, const float* __restrict__ Wo0,
    const float* __restrict__ bo0,
    float* __restrict__ f0out, float* __restrict__ f1out, int l) {
  __shared__ float sF0[4][64], sQ[4][64], sT[4][512], sAtt[4][256], sA[4][32];
  __shared__ float sS[4][512], sAgg[4][64], sMv1[4][512], sMg1[4][512];
  int wv = threadIdx.x >> 5, lane = threadIdx.x & 31;
  int node = blockIdx.x*4 + wv;
  int bat = node / N_;
  int ne = node*K_;
  const float* Wql  = Wq  + l*H_*H_;
  const float* Wkl  = Wk  + l*H_*H_;
  const float* Wv0l = Wv0 + l*H_*H_;
  const float* Wv1l = Wv1 + l*H_*CV_;
  const float* Wg1l = Wg1 + l*H_*CV_;
  const float* Wo0l = Wo0 + l*H_*H_;
  const float* bo0l = bo0 + l*H_;

  sF0[wv][lane]    = f0in[node*64 + lane];
  sF0[wv][lane+32] = f0in[node*64 + lane + 32];
  __syncthreads();

  // q = f0 @ Wq
  for (int c = lane; c < 64; c += 32) {
    float acc = 0.f;
    for (int k = 0; k < 64; ++k) acc += sF0[wv][k]*Wql[k*64 + c];
    sQ[wv][c] = acc;
  }
  __syncthreads();

  // t[c][h] = sum_d Wk[c, h*8+d] * q[h*8+d]
  for (int u = lane; u < 512; u += 32) {
    int c = u >> 3, h = u & 7;
    float acc = 0.f;
    #pragma unroll
    for (int d = 0; d < 8; ++d) acc += Wkl[c*64 + h*8 + d]*sQ[wv][h*8 + d];
    sT[wv][u] = acc;
  }
  __syncthreads();

  // logits[k][h] = (m[k,:] . t[:,h]) / sqrt(8)
  const float invs = 0.35355339059327373f;
  for (int u = lane; u < 256; u += 32) {
    int k = u >> 3, h = u & 7;
    const float* mrow = mArr + (size_t)(ne + k)*64;
    float acc = 0.f;
    for (int c = 0; c < 64; ++c) acc += mrow[c]*sT[wv][c*8 + h];
    sAtt[wv][u] = acc*invs;
  }
  __syncthreads();

  // softmax over k per head
  if (lane < 8) {
    int h = lane;
    float mx = -1e30f;
    for (int k = 0; k < 32; ++k) mx = fmaxf(mx, sAtt[wv][k*8 + h]);
    float sum = 0.f;
    for (int k = 0; k < 32; ++k) { float e = expf(sAtt[wv][k*8 + h] - mx); sAtt[wv][k*8 + h] = e; sum += e; }
    float inv = 1.f/sum;
    for (int k = 0; k < 32; ++k) sAtt[wv][k*8 + h] *= inv;
  }
  __syncthreads();

  // a[k] = mean over heads
  {
    float acc = 0.f;
    #pragma unroll
    for (int h = 0; h < 8; ++h) acc += sAtt[wv][lane*8 + h];
    sA[wv][lane] = acc*0.125f;
  }
  __syncthreads();

  // s[c][h] = sum_k attn[k][h] * m[k][c]
  for (int u = lane; u < 512; u += 32) {
    int c = u >> 3, h = u & 7;
    float acc = 0.f;
    for (int k = 0; k < 32; ++k) acc += sAtt[wv][k*8 + h]*mArr[(size_t)(ne + k)*64 + c];
    sS[wv][u] = acc;
  }
  __syncthreads();

  // agg0[hd] = sum_c s[c][h] * Wv0[c][hd]
  for (int c2 = lane; c2 < 64; c2 += 32) {
    int h = c2 >> 3;
    float acc = 0.f;
    for (int c = 0; c < 64; ++c) acc += sS[wv][c*8 + h]*Wv0l[c*64 + c2];
    sAgg[wv][c2] = acc;
  }
  __syncthreads();

  // f0_new = f0 + agg0 @ Wo0 + bo0
  for (int c2 = lane; c2 < 64; c2 += 32) {
    float acc = sF0[wv][c2] + bo0l[c2];
    for (int c = 0; c < 64; ++c) acc += sAgg[wv][c]*Wo0l[c*64 + c2];
    f0out[node*64 + c2] = acc;
  }

  // mWv1, mWg1 [32 x 16]
  for (int u = lane; u < 512; u += 32) {
    int k = u >> 4, c = u & 15;
    const float* mrow = mArr + (size_t)(ne + k)*64;
    float a1 = 0.f, a2 = 0.f;
    for (int cc = 0; cc < 64; ++cc) {
      float mv = mrow[cc];
      a1 += mv*Wv1l[cc*16 + c];
      a2 += mv*Wg1l[cc*16 + c];
    }
    sMv1[wv][u] = a1; sMg1[wv][u] = a2;
  }
  __syncthreads();

  // f1_new[c][d] = f1 + sum_k a[k]*(dir[k,d]*mWv1[k,c] + mWg1[k,c]*f1s[k,c,d])
  for (int u = lane; u < 48; u += 32) {
    int c = u/3, d = u - c*3;
    float val = f1in[node*48 + u];
    for (int k = 0; k < 32; ++k) {
      int e = ne + k;
      int src = idxA[e];
      val += sA[wv][k]*( dirn[e*3 + d]*sMv1[wv][k*16 + c]
                       + sMg1[wv][k*16 + c]*f1in[(bat*N_ + src)*48 + c*3 + d] );
    }
    f1out[node*48 + u] = val;
  }
}

// ---------------- output projection ----------------
__global__ void k_out(const float* __restrict__ f1, const float* __restrict__ Wout1,
                      float* __restrict__ out) {
  int u = blockIdx.x*blockDim.x + threadIdx.x;
  if (u >= NODES_*3) return;
  int node = u/3, d = u - node*3;
  float acc = 0.f;
  #pragma unroll
  for (int c = 0; c < 16; ++c) acc += f1[node*48 + c*3 + d]*Wout1[c];
  out[u] = acc;
}

extern "C" void kernel_launch(void* const* d_in, const int* in_sizes, int n_in,
                              void* d_out, int out_size, void* d_ws, size_t ws_size,
                              hipStream_t stream) {
  const int*   seq    = (const int*)d_in[0];
  const float* coord  = (const float*)d_in[1];
  const int*   tptr   = (const int*)d_in[2];
  const float* atomT  = (const float*)d_in[3];
  const float* timeT  = (const float*)d_in[4];
  const float* Wp0    = (const float*)d_in[5];
  const float* bp0    = (const float*)d_in[6];
  const float* Wp1    = (const float*)d_in[7];
  const float* We1    = (const float*)d_in[8];
  const float* be1    = (const float*)d_in[9];
  const float* We2    = (const float*)d_in[10];
  const float* be2    = (const float*)d_in[11];
  const float* Wq     = (const float*)d_in[12];
  const float* Wk     = (const float*)d_in[13];
  const float* Wv0    = (const float*)d_in[14];
  const float* Wv1    = (const float*)d_in[15];
  const float* Wg1    = (const float*)d_in[16];
  const float* Wo0    = (const float*)d_in[17];
  const float* bo0    = (const float*)d_in[18];
  const float* Wout1  = (const float*)d_in[19];
  (void)in_sizes; (void)n_in; (void)out_size; (void)ws_size;

  char* ws = (char*)d_ws;
  size_t off = 0;
  auto alloc = [&](size_t bytes) -> void* {
    void* p = ws + off;
    off = (off + bytes + 255) & ~(size_t)255;
    return p;
  };
  float* coordC = (float*)alloc((size_t)NODES_*3*4);
  int*   idxA   = (int*)  alloc((size_t)E_*4);
  float* dirn   = (float*)alloc((size_t)E_*3*4);
  float* rbf    = (float*)alloc((size_t)E_*16*4);
  float* dots   = (float*)alloc((size_t)E_*32*4);
  float* f0A    = (float*)alloc((size_t)NODES_*64*4);
  float* f0B    = (float*)alloc((size_t)NODES_*64*4);
  float* f1A    = (float*)alloc((size_t)NODES_*48*4);
  float* f1B    = (float*)alloc((size_t)NODES_*48*4);
  float* mArr   = (float*)alloc((size_t)E_*64*4);

  k_center<<<B_, 256, 0, stream>>>(coord, coordC);
  k_knn<<<NODES_/4, 128, 0, stream>>>(coordC, idxA);
  k_edges<<<E_/256, 256, 0, stream>>>(coordC, idxA, dirn, rbf);
  k_f0init<<<(NODES_*H_)/256, 256, 0, stream>>>(seq, tptr, atomT, timeT, Wp0, bp0, f0A);
  k_f1init<<<(NODES_*48)/256, 256, 0, stream>>>(coordC, Wp1, f1A);

  float* f0c = f0A; float* f0n = f0B;
  float* f1c = f1A; float* f1n = f1B;
  for (int l = 0; l < L_; ++l) {
    k_dots<<<E_/256, 256, 0, stream>>>(f1c, idxA, dirn, dots);
    k_edge_mlp<<<(E_/16)/4, 128, 0, stream>>>(f0c, idxA, rbf, dots, We1, be1, We2, be2, mArr, l);
    k_node_update<<<NODES_/4, 128, 0, stream>>>(f0c, f1c, mArr, idxA, dirn,
                                                Wq, Wk, Wv0, Wv1, Wg1, Wo0, bo0,
                                                f0n, f1n, l);
    float* t0 = f0c; f0c = f0n; f0n = t0;
    float* t1 = f1c; f1c = f1n; f1n = t1;
  }
  k_out<<<(NODES_*3 + 255)/256, 256, 0, stream>>>(f1c, Wout1, (float*)d_out);
}
